// BlueBoxLayer_59021440582409
// MI455X (gfx1250) — matrix operand order repaired and checked by reading, not compiled, not run on hardware
//
#include <hip/hip_runtime.h>

typedef __bf16 bf16;
typedef __attribute__((ext_vector_type(8)))  __bf16 v8bf;
typedef __attribute__((ext_vector_type(16))) __bf16 v16bf;
typedef __attribute__((ext_vector_type(8)))  float  v8f;

#define D_DIM 128
#define T_DIM 8192
#define N_DIM 512
#define M_DIM 384

// ---- WMMA tile loaders (bf16 16x16x32, wave32; A/B use this code's contiguous K order) ----------------
// A (16x32, MxK): lane L holds row M=L%16; halves e=0..15 -> K=base+e,
// base = 16*(L/16).  => two contiguous 16B loads at
// p and p+8 (p already includes +base).
static __device__ __forceinline__ v16bf load_a_tile(const bf16* p) {
    union { v16bf v; v8bf h[2]; } u;
    u.h[0] = *reinterpret_cast<const v8bf*>(p);
    u.h[1] = *reinterpret_cast<const v8bf*>(p + 8);
    return u.v;
}
// B (32x16, KxN): lane L holds col N=L%16; halves e=0..15 -> K = 16*(L/16)+e,
// i.e. 16 contiguous K values (p already includes +kbase).
static __device__ __forceinline__ v16bf load_b_tile(const bf16* p) {
    union { v16bf v; v8bf h[2]; } u;
    u.h[0] = *reinterpret_cast<const v8bf*>(p);
    u.h[1] = *reinterpret_cast<const v8bf*>(p + 8);
    return u.v;
}
// C (16x16 f32): reg r, lane L -> M = r + 8*(L/16), N = L%16.

// ---- 1) DH = D@H in fp32, stored transposed as bf16 DHt[T][512] ------------
__global__ void k_dh(const float* __restrict__ D, const float* __restrict__ H,
                     bf16* __restrict__ DHt) {
    const int t = blockIdx.x;                       // uniform per block -> scalar H loads
    const int n = blockIdx.y * blockDim.x + threadIdx.x;
    float acc = 0.f;
    const float* Dr = D + n * D_DIM;
    #pragma unroll 8
    for (int d = 0; d < D_DIM; ++d)
        acc += Dr[d] * H[d * T_DIM + t];
    DHt[t * N_DIM + n] = (bf16)acc;                 // coalesced over n
}

// ---- 2) log_beta[t] = -0.5 * sum_n DH[n][t]^2 (from bf16, per-lane contiguous)
__global__ void k_logbeta(const bf16* __restrict__ DHt, float* __restrict__ lb) {
    const int t = blockIdx.x * blockDim.x + threadIdx.x;
    const bf16* row = DHt + t * N_DIM;
    float acc = 0.f;
    #pragma unroll 8
    for (int n = 0; n < N_DIM; ++n) { float v = (float)row[n]; acc += v * v; }
    lb[t] = -0.5f * acc;
}

// ---- 3) Hb = bf16(H) --------------------------------------------------------
__global__ void k_hb(const float* __restrict__ H, bf16* __restrict__ Hb) {
    const int i = blockIdx.x * blockDim.x + threadIdx.x;
    Hb[i] = (bf16)H[i];
}

// ---- 4) Pass 1: per-row softmax stats m[t], L[t] over S[t,j]=G[t,j]+lb[j] ---
// 4 waves/block, each wave owns 16 t-rows; A tiles (K=512) resident in VGPRs.
__global__ void __launch_bounds__(128)
k_stats(const bf16* __restrict__ DHt, const float* __restrict__ lb,
        float* __restrict__ mrow, float* __restrict__ lrow) {
    const int lane = threadIdx.x & 31;
    const int wave = threadIdx.x >> 5;
    const int t0 = blockIdx.x * 64 + wave * 16;
    const int lm = lane & 15, lh = lane >> 4;

    v16bf aT[16];
    #pragma unroll
    for (int kc = 0; kc < 16; ++kc)
        aT[kc] = load_a_tile(DHt + (t0 + lm) * N_DIM + kc * 32 + lh * 16);

    float m[8], l[8];
    #pragma unroll
    for (int r = 0; r < 8; ++r) { m[r] = -3.0e38f; l[r] = 0.f; }

    for (int j0 = 0; j0 < T_DIM; j0 += 16) {
        const bf16* pb = DHt + (j0 + lm) * N_DIM + lh * 16;
        v8f c = {};
        #pragma unroll
        for (int kc = 0; kc < 16; ++kc) {
            v16bf b = load_b_tile(pb + kc * 32);
            c = __builtin_amdgcn_wmma_f32_16x16x32_bf16(false, aT[kc], false, b,
                                                        (short)0, c, false, false);
        }
        const float bj = lb[j0 + lm];
        #pragma unroll
        for (int r = 0; r < 8; ++r) {
            float s  = c[r] + bj;
            float nm = fmaxf(m[r], s);
            l[r] = l[r] * __expf(m[r] - nm) + __expf(s - nm);
            m[r] = nm;
        }
    }
    // reduce (m,l) across the 16 lanes holding the same row
    #pragma unroll
    for (int r = 0; r < 8; ++r) {
        #pragma unroll
        for (int off = 1; off < 16; off <<= 1) {
            float om = __shfl_xor(m[r], off, 32);
            float ol = __shfl_xor(l[r], off, 32);
            float nm = fmaxf(m[r], om);
            l[r] = l[r] * __expf(m[r] - nm) + ol * __expf(om - nm);
            m[r] = nm;
        }
        if (lm == 0) {
            const int row = t0 + r + lh * 8;
            mrow[row] = m[r];
            lrow[row] = l[r];
        }
    }
}

// ---- 5) Pass 2: Z[:,j] = l2 * sum_t H[:,t] * exp(S[t,j]-m[t]) / L[t] --------
// 4 waves/block; each wave owns 16 j-cols (B_G resident), streams t in 32s,
// stages P through LDS into B layout, accumulates Z as 8 resident C tiles.
__global__ void __launch_bounds__(128)
k_zpass(const bf16* __restrict__ DHt, const bf16* __restrict__ Hb,
        const float* __restrict__ lb, const float* __restrict__ mrow,
        const float* __restrict__ lrow, const float* __restrict__ l2p,
        float* __restrict__ Z) {
    __shared__ __align__(16) bf16 Plds[4][16 * 32];   // per-wave [n][k] staging
    const int lane = threadIdx.x & 31;
    const int wave = threadIdx.x >> 5;
    const int j0 = blockIdx.x * 64 + wave * 16;
    const int lm = lane & 15, lh = lane >> 4;

    v16bf bj[16];
    #pragma unroll
    for (int kc = 0; kc < 16; ++kc)
        bj[kc] = load_b_tile(DHt + (j0 + lm) * N_DIM + kc * 32 + lh * 16);

    const float bjlb = lb[j0 + lm];
    const v8f zinit = {};
    v8f zc[8];
    #pragma unroll
    for (int d = 0; d < 8; ++d) zc[d] = zinit;

    for (int tt = 0; tt < T_DIM; tt += 32) {
        #pragma unroll
        for (int g = 0; g < 2; ++g) {
            const int tb = tt + g * 16;
            v8f gc = {};
            #pragma unroll
            for (int kc = 0; kc < 16; ++kc) {
                v16bf a = load_a_tile(DHt + (tb + lm) * N_DIM + kc * 32 + lh * 16);
                gc = __builtin_amdgcn_wmma_f32_16x16x32_bf16(false, a, false, bj[kc],
                                                             (short)0, gc, false, false);
            }
            #pragma unroll
            for (int r = 0; r < 8; ++r) {
                const int trow = tb + r + lh * 8;
                float p = __expf(gc[r] + bjlb - mrow[trow]) / lrow[trow];
                Plds[wave][lm * 32 + g * 16 + r + lh * 8] = (bf16)p;
            }
        }
        __syncthreads();
        v16bf bp = load_b_tile(&Plds[wave][lm * 32 + lh * 16]);
        __syncthreads();
        #pragma unroll
        for (int d = 0; d < 8; ++d) {
            v16bf a = load_a_tile(Hb + (d * 16 + lm) * T_DIM + tt + lh * 16);
            zc[d] = __builtin_amdgcn_wmma_f32_16x16x32_bf16(false, a, false, bp,
                                                            (short)0, zc[d], false, false);
        }
    }
    const float l2 = l2p[0];
    #pragma unroll
    for (int d = 0; d < 8; ++d)
        #pragma unroll
        for (int r = 0; r < 8; ++r)
            Z[(d * 16 + r + lh * 8) * T_DIM + j0 + lm] = l2 * zc[d][r];
}

// ---- 6) out = softthresh(U@Z + V@X, l1) ------------------------------------
__global__ void k_final(const float* __restrict__ U, const float* __restrict__ V,
                        const float* __restrict__ Z, const float* __restrict__ X,
                        const float* __restrict__ l1p, float* __restrict__ out) {
    const int t = blockIdx.x * blockDim.x + threadIdx.x;
    const int d = blockIdx.y;
    float acc = 0.f;
    const float* Ur = U + d * D_DIM;
    #pragma unroll 4
    for (int k = 0; k < D_DIM; ++k) acc += Ur[k] * Z[k * T_DIM + t];
    const float* Vr = V + d * M_DIM;
    #pragma unroll 4
    for (int k = 0; k < M_DIM; ++k) acc += Vr[k] * X[k * T_DIM + t];
    const float th = l1p[0];                        // thres = l1 / C_INIT, C_INIT = 1
    float r = fmaxf(fabsf(acc) - th, 0.f);
    out[d * T_DIM + t] = copysignf(r, acc);
}

extern "C" void kernel_launch(void* const* d_in, const int* in_sizes, int n_in,
                              void* d_out, int out_size, void* d_ws, size_t ws_size,
                              hipStream_t stream) {
    const float* H  = (const float*)d_in[0];
    const float* D  = (const float*)d_in[1];
    const float* X  = (const float*)d_in[2];
    const float* U  = (const float*)d_in[3];
    const float* V  = (const float*)d_in[4];
    const float* l1 = (const float*)d_in[5];
    const float* l2 = (const float*)d_in[6];
    float* out = (float*)d_out;

    char* ws = (char*)d_ws;
    bf16*  DHt  = (bf16*)ws;  ws += (size_t)T_DIM * N_DIM * sizeof(bf16);   // 8 MB
    bf16*  Hb   = (bf16*)ws;  ws += (size_t)D_DIM * T_DIM * sizeof(bf16);   // 2 MB
    float* lb   = (float*)ws; ws += (size_t)T_DIM * sizeof(float);
    float* mrow = (float*)ws; ws += (size_t)T_DIM * sizeof(float);
    float* lrow = (float*)ws; ws += (size_t)T_DIM * sizeof(float);
    float* Z    = (float*)ws; ws += (size_t)D_DIM * T_DIM * sizeof(float);  // 4 MB

    k_dh<<<dim3(T_DIM, N_DIM / 256), 256, 0, stream>>>(D, H, DHt);
    k_logbeta<<<dim3(T_DIM / 256), 256, 0, stream>>>(DHt, lb);
    k_hb<<<dim3(D_DIM * T_DIM / 256), 256, 0, stream>>>(H, Hb);
    k_stats<<<dim3(T_DIM / 64), 128, 0, stream>>>(DHt, lb, mrow, lrow);
    k_zpass<<<dim3(T_DIM / 64), 128, 0, stream>>>(DHt, Hb, lb, mrow, lrow, l2, Z);
    k_final<<<dim3(T_DIM / 256, D_DIM), 256, 0, stream>>>(U, V, Z, X, l1, out);
}
